// GCN_84413287236342
// MI455X (gfx1250) — compile-verified, matrix operand order checked by reading, not run on hardware
//
#include <hip/hip_runtime.h>

// ---------------------------------------------------------------------------
// GCN forward for MI455X (gfx1250):
//   - dense GEMMs: bf16 WMMA (v_wmma_f32_16x16x32_bf16), f32 accumulate,
//     A-tile staged into LDS via global_load_async_to_lds_b128 (inline asm)
//   - SpMM: bf16 gather (half traffic) + f32 global atomic scatter
// Output tuple: (out[64,128], h[100000,128]) concatenated flat in d_out.
// ---------------------------------------------------------------------------

typedef __attribute__((ext_vector_type(16))) __bf16 v16bf;
typedef __attribute__((ext_vector_type(8)))  __bf16 v8bf;
typedef __attribute__((ext_vector_type(4)))  __bf16 v4bf;
typedef __attribute__((ext_vector_type(8)))  float  v8f;

#define D_FEAT 128
#define N_GRAPHS_C 64
#define POOL_CHUNK 256
#define LDS_STRIDE 136   // 128 + 8 bf16 pad -> row stride 272B, bank-conflict free
#define USE_ASYNC_LDS 1  // gfx1250 async-to-LDS staging via inline asm

// -------------------- generic zero (grid-stride) ---------------------------
__global__ void __launch_bounds__(256) zero_f32_kernel(float* __restrict__ p, size_t n) {
    size_t i = (size_t)blockIdx.x * blockDim.x + threadIdx.x;
    size_t stride = (size_t)gridDim.x * blockDim.x;
    for (; i < n; i += stride) p[i] = 0.0f;
}

// -------------------- degree accumulation ----------------------------------
__global__ void __launch_bounds__(256) deg_kernel(const int* __restrict__ src,
                                                  const int* __restrict__ dst,
                                                  float* __restrict__ deg_out,
                                                  float* __restrict__ deg_in,
                                                  int n_edges) {
    int e = blockIdx.x * 256 + threadIdx.x;
    if (e >= n_edges) return;
    atomicAdd(&deg_out[src[e]], 1.0f);
    atomicAdd(&deg_in[dst[e]], 1.0f);
}

// norm = clip(deg,1)^-0.5, in place on both arrays
__global__ void __launch_bounds__(256) norm_kernel(float* __restrict__ a,
                                                   float* __restrict__ b,
                                                   int n) {
    int i = blockIdx.x * 256 + threadIdx.x;
    if (i >= n) return;
    a[i] = rsqrtf(fmaxf(a[i], 1.0f));
    b[i] = rsqrtf(fmaxf(b[i], 1.0f));
}

// -------------------- weight repack: f32 row-major -> bf16 B-fragment ------
// B-fragment layout (16-bit 32x16 B matrix, wave32):
//   lane (n = lane&15, hi = lane>>4) element e holds
//   k = (e<8) ? hi*8+e : 16+hi*8+(e-8),  col = nt*16 + n, row = kt*32 + k.
// Stored as WB[((kt*8+nt)*32 + lane)*16 + e] so GEMM does one v16bf load/lane.
__global__ void __launch_bounds__(256) prep_w_kernel(const float* __restrict__ W,
                                                     __bf16* __restrict__ WB) {
    int i = blockIdx.x * 256 + threadIdx.x;          // 0 .. 4*8*32*16-1
    if (i >= 4 * 8 * 32 * 16) return;
    int e    = i & 15;
    int lane = (i >> 4) & 31;
    int nt   = (i >> 9) & 7;
    int kt   = i >> 12;
    int hi = lane >> 4;
    int n  = lane & 15;
    int k  = (e < 8) ? (hi * 8 + e) : (16 + hi * 8 + (e - 8));
    WB[i] = (__bf16)W[(kt * 32 + k) * D_FEAT + nt * 16 + n];
}

// xb = bf16(x * norm_src[node])
__global__ void __launch_bounds__(256) prep_x_kernel(const float* __restrict__ x,
                                                     const float* __restrict__ norm_src,
                                                     __bf16* __restrict__ xb,
                                                     size_t total) {
    size_t i = (size_t)blockIdx.x * 256 + threadIdx.x;
    if (i >= total) return;
    xb[i] = (__bf16)(x[i] * norm_src[i >> 7]);
}

// -------------------- dense GEMM: C[N,128] = A[N,128] @ W[128,128] ---------
// 1 block = 8 waves; all waves share one 16-row A tile staged in LDS via the
// gfx1250 async-to-LDS path; wave nt computes the 16x16 tile
// (rowtile=blockIdx.x, coltile=nt) with 4 WMMAs.
// Grid is exact (N % 16 == 0) -> no divergence, EXEC all-1s around WMMA.
__global__ void __launch_bounds__(256) gemm_bf16_kernel(const __bf16* __restrict__ A,
                                                        const __bf16* __restrict__ WB,
                                                        __bf16* __restrict__ C) {
    __shared__ __bf16 ldsA[16 * LDS_STRIDE];

    const int lane = threadIdx.x & 31;
    const int nt   = threadIdx.x >> 5;   // column tile 0..7
    const int rt   = blockIdx.x;         // row tile
    const int m    = lane & 15;
    const int hi   = lane >> 4;

    // ---- stage A tile: 16 rows x 128 bf16 = 4KB, 16B chunk per thread ----
    {
        const int t = threadIdx.x;
        const int r = t >> 4;            // row 0..15
        const int c = t & 15;            // 16B chunk 0..15
        const __bf16* gp = A + ((size_t)rt * 16 + r) * D_FEAT + c * 8;
        __bf16* lp = &ldsA[r * LDS_STRIDE + c * 8];
#if USE_ASYNC_LDS
        // Generic LDS pointer: low 32 bits are the LDS byte address (ISA 10.2,
        // LDS aperture).
        unsigned lds_off = (unsigned)(unsigned long long)lp;
        asm volatile("global_load_async_to_lds_b128 %0, %1, off"
                     :
                     : "v"(lds_off), "v"(gp)
                     : "memory");
        asm volatile("s_wait_asynccnt 0x0" ::: "memory");
#else
        *(v8bf*)lp = *(const v8bf*)gp;
#endif
    }
    __syncthreads();

    const __bf16* lrow = &ldsA[m * LDS_STRIDE];

    v8f acc = {};
#pragma unroll
    for (int kt = 0; kt < 4; ++kt) {
        // A fragment from LDS (16-bit 16x32 A layout: two 8-elem runs)
        v8bf alo = *(const v8bf*)(lrow + kt * 32 + hi * 8);
        v8bf ahi = *(const v8bf*)(lrow + kt * 32 + 16 + hi * 8);
        v16bf a;
#pragma unroll
        for (int i = 0; i < 8; ++i) { a[i] = alo[i]; a[i + 8] = ahi[i]; }
        // B fragment: prepacked, one contiguous 32B load per lane
        v16bf b = *(const v16bf*)(WB + (((size_t)kt * 8 + nt) * 32 + lane) * 16);
        acc = __builtin_amdgcn_wmma_f32_16x16x32_bf16(
            /*neg_a=*/false, a, /*neg_b=*/false, b,
            /*c_mod=*/(short)0, acc, /*reuse_a=*/false, /*reuse_b=*/false);
    }
    // D layout: VGPR r -> row hi*8+r, col m.  Store bf16 (halves SpMM gather).
    __bf16* out = C + ((size_t)rt * 16 + hi * 8) * D_FEAT + nt * 16 + m;
#pragma unroll
    for (int r = 0; r < 8; ++r) out[(size_t)r * D_FEAT] = (__bf16)acc[r];
}

// -------------------- SpMM scatter: AGG[dst] += C[src] ---------------------
// One wave per edge; each lane gathers 4 bf16 features (8B) and issues
// 4 f32 atomic adds -> accumulation stays f32.
__global__ void __launch_bounds__(256) scatter_kernel(const __bf16* __restrict__ C,
                                                      const int* __restrict__ src,
                                                      const int* __restrict__ dst,
                                                      float* __restrict__ AGG,
                                                      int n_edges) {
    int e = blockIdx.x * 8 + (threadIdx.x >> 5);
    if (e >= n_edges) return;
    int lane = threadIdx.x & 31;
    int s = src[e];
    int d = dst[e];
    const v4bf v = *(const v4bf*)(C + (size_t)s * D_FEAT + lane * 4);
    float* p = AGG + (size_t)d * D_FEAT + lane * 4;
    atomicAdd(p + 0, (float)v[0]);
    atomicAdd(p + 1, (float)v[1]);
    atomicAdd(p + 2, (float)v[2]);
    atomicAdd(p + 3, (float)v[3]);
}

// -------------------- post layer 1: relu + rescale -> bf16 layer-2 input ---
__global__ void __launch_bounds__(256) post1_kernel(const float* __restrict__ AGG,
                                                    const float* __restrict__ norm_dst,
                                                    const float* __restrict__ bias,
                                                    const float* __restrict__ norm_src,
                                                    __bf16* __restrict__ xb2,
                                                    size_t total) {
    size_t i = (size_t)blockIdx.x * 256 + threadIdx.x;
    if (i >= total) return;
    size_t node = i >> 7;
    int f = (int)(i & 127);
    float v = AGG[i] * norm_dst[node] + bias[f];
    xb2[i] = (__bf16)(fmaxf(v, 0.0f) * norm_src[node]);
}

// -------------------- post layer 2: relu -> f32 h output -------------------
__global__ void __launch_bounds__(256) post2_kernel(const float* __restrict__ AGG,
                                                    const float* __restrict__ norm_dst,
                                                    const float* __restrict__ bias,
                                                    float* __restrict__ h_out,
                                                    size_t total) {
    size_t i = (size_t)blockIdx.x * 256 + threadIdx.x;
    if (i >= total) return;
    size_t node = i >> 7;
    int f = (int)(i & 127);
    float v = AGG[i] * norm_dst[node] + bias[f];
    h_out[i] = fmaxf(v, 0.0f);
}

// -------------------- segment-mean pooling (graph_id sorted) ---------------
// 128 threads = feature dim; block handles POOL_CHUNK consecutive nodes,
// flushing a register accumulator on each graph transition (few atomics).
__global__ void __launch_bounds__(128) pool_kernel(const float* __restrict__ h,
                                                   const int* __restrict__ gid,
                                                   float* __restrict__ sums,
                                                   float* __restrict__ counts,
                                                   int n_nodes) {
    int f  = threadIdx.x;
    int n0 = blockIdx.x * POOL_CHUNK;
    if (n0 >= n_nodes) return;
    int n1 = n0 + POOL_CHUNK;
    if (n1 > n_nodes) n1 = n_nodes;
    int cur = gid[n0];
    float acc = 0.0f, cnt = 0.0f;
    for (int n = n0; n < n1; ++n) {
        int g = gid[n];
        if (g != cur) {
            atomicAdd(&sums[(size_t)cur * D_FEAT + f], acc);
            if (f == 0) atomicAdd(&counts[cur], cnt);
            acc = 0.0f; cnt = 0.0f; cur = g;
        }
        acc += h[(size_t)n * D_FEAT + f];
        cnt += 1.0f;
    }
    atomicAdd(&sums[(size_t)cur * D_FEAT + f], acc);
    if (f == 0) atomicAdd(&counts[cur], cnt);
}

__global__ void __launch_bounds__(256) pool_div_kernel(const float* __restrict__ sums,
                                                       const float* __restrict__ counts,
                                                       float* __restrict__ hg) {
    int i = blockIdx.x * 256 + threadIdx.x;
    if (i >= N_GRAPHS_C * D_FEAT) return;
    hg[i] = sums[i] / fmaxf(counts[i >> 7], 1.0f);
}

// -------------------- tiny readout MLP (64x128 @ 128x128), exact f32 -------
__global__ void __launch_bounds__(128) readout_kernel(const float* __restrict__ in,
                                                      const float* __restrict__ W,
                                                      const float* __restrict__ bias,
                                                      float* __restrict__ out,
                                                      int do_relu) {
    int g = blockIdx.x;   // graph
    int n = threadIdx.x;  // output feature
    float acc = bias[n];
#pragma unroll 8
    for (int k = 0; k < D_FEAT; ++k)
        acc = fmaf(in[g * D_FEAT + k], W[k * D_FEAT + n], acc);
    if (do_relu) acc = fmaxf(acc, 0.0f);
    out[g * D_FEAT + n] = acc;
}

// ---------------------------------------------------------------------------
extern "C" void kernel_launch(void* const* d_in, const int* in_sizes, int n_in,
                              void* d_out, int out_size, void* d_ws, size_t ws_size,
                              hipStream_t stream) {
    const float* x    = (const float*)d_in[0];
    const int*   ei   = (const int*)d_in[1];
    const int*   gid  = (const int*)d_in[2];
    // d_in[3] = num_graphs scalar (compile-time 64 in reference)
    const float* W1   = (const float*)d_in[4];
    const float* b1   = (const float*)d_in[5];
    const float* W2   = (const float*)d_in[6];
    const float* b2   = (const float*)d_in[7];
    const float* Wr1  = (const float*)d_in[8];
    const float* br1  = (const float*)d_in[9];
    const float* Wr2  = (const float*)d_in[10];
    const float* br2  = (const float*)d_in[11];

    const int n_nodes = in_sizes[0] / D_FEAT;   // 100000
    const int n_edges = in_sizes[1] / 2;        // 1600000
    const int* src = ei;
    const int* dst = ei + n_edges;

    const size_t NF = (size_t)n_nodes * D_FEAT;

    // ---- carve workspace (256B aligned chunks) ----
    char*  ws  = (char*)d_ws;
    size_t off = 0;
    auto carve = [&](size_t bytes) -> char* {
        off = (off + 255) & ~(size_t)255;
        char* p = ws + off;
        off += bytes;
        return p;
    };
    float*  norm_src = (float*)carve((size_t)n_nodes * 4);
    float*  norm_dst = (float*)carve((size_t)n_nodes * 4);
    __bf16* xb       = (__bf16*)carve(NF * 2);          // bf16 GEMM input (both layers)
    __bf16* Cbuf     = (__bf16*)carve(NF * 2);          // projected features (bf16 gather)
    float*  AGG      = (float*)carve(NF * 4);           // f32 scatter accumulator
    __bf16* W1B      = (__bf16*)carve(4 * 8 * 32 * 16 * 2);
    __bf16* W2B      = (__bf16*)carve(4 * 8 * 32 * 16 * 2);
    float*  sums     = (float*)carve((size_t)N_GRAPHS_C * D_FEAT * 4);
    float*  counts   = (float*)carve((size_t)N_GRAPHS_C * 4);
    float*  hg       = (float*)carve((size_t)N_GRAPHS_C * D_FEAT * 4);
    float*  zbuf     = (float*)carve((size_t)N_GRAPHS_C * D_FEAT * 4);

    float* out_mlp = (float*)d_out;                        // (64,128)
    float* h_out   = (float*)d_out + N_GRAPHS_C * D_FEAT;  // (100000,128)

    // ---- degrees & norms ----
    zero_f32_kernel<<<512, 256, 0, stream>>>(norm_src, (size_t)n_nodes);
    zero_f32_kernel<<<512, 256, 0, stream>>>(norm_dst, (size_t)n_nodes);
    zero_f32_kernel<<<64, 256, 0, stream>>>(sums, (size_t)N_GRAPHS_C * D_FEAT);
    zero_f32_kernel<<<1, 256, 0, stream>>>(counts, (size_t)N_GRAPHS_C);
    deg_kernel<<<(n_edges + 255) / 256, 256, 0, stream>>>(src, dst, norm_src, norm_dst, n_edges);
    norm_kernel<<<(n_nodes + 255) / 256, 256, 0, stream>>>(norm_src, norm_dst, n_nodes);

    // ---- weight repack + input prep ----
    prep_w_kernel<<<(4 * 8 * 32 * 16 + 255) / 256, 256, 0, stream>>>(W1, W1B);
    prep_w_kernel<<<(4 * 8 * 32 * 16 + 255) / 256, 256, 0, stream>>>(W2, W2B);
    prep_x_kernel<<<(unsigned)((NF + 255) / 256), 256, 0, stream>>>(x, norm_src, xb, NF);

    const unsigned gemm_grid    = (unsigned)(n_nodes / 16);          // 6250, exact
    const unsigned scatter_grid = (unsigned)((n_edges + 7) / 8);

    // ---- layer 1 ----
    gemm_bf16_kernel<<<gemm_grid, 256, 0, stream>>>(xb, W1B, Cbuf);
    zero_f32_kernel<<<2048, 256, 0, stream>>>(AGG, NF);
    scatter_kernel<<<scatter_grid, 256, 0, stream>>>(Cbuf, src, dst, AGG, n_edges);
    post1_kernel<<<(unsigned)((NF + 255) / 256), 256, 0, stream>>>(AGG, norm_dst, b1, norm_src, xb, NF);

    // ---- layer 2 ----
    gemm_bf16_kernel<<<gemm_grid, 256, 0, stream>>>(xb, W2B, Cbuf);
    zero_f32_kernel<<<2048, 256, 0, stream>>>(AGG, NF);
    scatter_kernel<<<scatter_grid, 256, 0, stream>>>(Cbuf, src, dst, AGG, n_edges);
    post2_kernel<<<(unsigned)((NF + 255) / 256), 256, 0, stream>>>(AGG, norm_dst, b2, h_out, NF);

    // ---- mean pooling + readout MLP ----
    pool_kernel<<<(n_nodes + POOL_CHUNK - 1) / POOL_CHUNK, 128, 0, stream>>>(h_out, gid, sums, counts, n_nodes);
    pool_div_kernel<<<(N_GRAPHS_C * D_FEAT + 255) / 256, 256, 0, stream>>>(sums, counts, hg);
    readout_kernel<<<N_GRAPHS_C, 128, 0, stream>>>(hg, Wr1, br1, zbuf, 1);
    readout_kernel<<<N_GRAPHS_C, 128, 0, stream>>>(zbuf, Wr2, br2, out_mlp, 0);
}
